// Attention_69879117906668
// MI455X (gfx1250) — matrix operand order repaired and checked by reading, not compiled, not run on hardware
//
#include <hip/hip_runtime.h>
#include <hip/hip_bf16.h>
#include <math.h>

typedef __attribute__((ext_vector_type(16))) _Float16 v16h;
typedef __attribute__((ext_vector_type(8)))  _Float16 v8h;
typedef __attribute__((ext_vector_type(8)))  float    v8f;

typedef unsigned int u32x4 __attribute__((ext_vector_type(4)));
typedef int          i32x4 __attribute__((ext_vector_type(4)));
typedef int          i32x8 __attribute__((ext_vector_type(8)));

#define NHID 512
#define TK   256
#define TQ   128
#define BB   8

#ifndef __has_builtin
#define __has_builtin(x) 0
#endif
#if __has_builtin(__builtin_amdgcn_tensor_load_to_lds) && __has_builtin(__builtin_amdgcn_s_wait_tensorcnt)
#define HAVE_TDM 1
#else
#define HAVE_TDM 0
#endif

#if HAVE_TDM
// 1-D contiguous TDM copy: global -> LDS, `words` DWORDs (D# per ISA 8.3/8.4:
// count=1, type=2(image), data_size=4B, tensor/tile_dim0=words, dim1=1).
__device__ __forceinline__ void tdm_load_1d(const void* gsrc, void* lds, unsigned words) {
  unsigned long long ga = (unsigned long long)(uintptr_t)gsrc;
  unsigned ldsaddr = (unsigned)(uintptr_t)lds;   // low 32 bits of generic ptr = LDS byte offset
  u32x4 g0;
  g0[0] = 1u;                                     // count=1, is_restore=0, gather off
  g0[1] = ldsaddr;                                // lds_addr
  g0[2] = (unsigned)ga;                           // global_addr[31:0]
  g0[3] = (unsigned)((ga >> 32) & 0x1FFFFFFull) | (2u << 30);  // global_addr[56:32] | type=2
  i32x8 g1 = (i32x8)0;
  g1[0] = (int)(2u << 16);                        // workgroup_mask=0, data_size=2 (4 bytes)
  g1[1] = (int)((words & 0xFFFFu) << 16);         // tensor_dim0[15:0] @ bits63:48
  g1[2] = (int)((words >> 16) & 0xFFFFu) | (1 << 16); // tensor_dim0[31:16]; tensor_dim1=1
  g1[3] = (int)((words & 0xFFFFu) << 16);         // tile_dim0 @ bits127:112
  g1[4] = 1;                                      // tile_dim1 = 1
  g1[5] = (int)words;                             // tensor_dim0_stride[31:0]
  i32x4 gz = (i32x4)0;
#if __clang_major__ >= 23
  i32x8 z8 = (i32x8)0;
  __builtin_amdgcn_tensor_load_to_lds(g0, g1, gz, gz, z8, 0);
#else
  __builtin_amdgcn_tensor_load_to_lds(g0, g1, gz, gz, 0);
#endif
}
#endif

// Cooperative fallback / generic panel load helper
__device__ __forceinline__ void coop_copy16(const void* gsrc, void* lds, int bytes) {
  const uint4* g = (const uint4*)gsrc;
  uint4* l = (uint4*)lds;
  for (int i = threadIdx.x; i < (bytes >> 4); i += blockDim.x) l[i] = g[i];
}

// ---------------------------------------------------------------------------
// Repack kernels: one-shot f32 -> f16 conversions into WMMA-friendly layouts.
// ---------------------------------------------------------------------------
__global__ __launch_bounds__(256) void cvt_f16_kernel(const float* __restrict__ src,
                                                      _Float16* __restrict__ dst, int n) {
  int i = blockIdx.x * 256 + threadIdx.x;
  if (i < n) dst[i] = (_Float16)src[i];
}

// W1 -> B-operand register image: Whp[((wh*32+t)*16+kk32)*32+lane][j] =
//   W1[wh*512 + kk32*32 + (lane>>4)*16 + j][t*16 + (lane&15)]
__global__ __launch_bounds__(256) void pack_w_kernel(const float* __restrict__ W1,
                                                     _Float16* __restrict__ Whp) {
  int o = blockIdx.x * 256 + threadIdx.x;          // 524288 total halves
  int j = o & 15, lane = (o >> 4) & 31, kk32 = (o >> 9) & 15, t = (o >> 13) & 31, wh = o >> 18;
  int krow = kk32 * 32 + (lane >> 4) * 16 + j;
  int col  = t * 16 + (lane & 15);
  Whp[o] = (_Float16)W1[(size_t)wh * NHID * NHID + (size_t)krow * NHID + col];
}

// c_key -> stage-3 B-operand image per batch:
// ckp[(((b*32+t)*8+kk32)*32+lane)*16+j] = c_key[(k*8+b)*512 + h], k=kk32*32+(lane>>4)*16+j, h=t*16+(lane&15)
__global__ __launch_bounds__(256) void pack_ck_kernel(const float* __restrict__ c_key,
                                                      _Float16* __restrict__ ckp) {
  int o = blockIdx.x * 256 + threadIdx.x;          // 1048576 total halves
  int j = o & 15, lane = (o >> 4) & 31, kk32 = (o >> 9) & 7, t = (o >> 12) & 31, b = o >> 17;
  int k = kk32 * 32 + (lane >> 4) * 16 + j;
  int h = t * 16 + (lane & 15);
  ckp[o] = (_Float16)c_key[((size_t)k * BB + b) * NHID + h];
}

// ---------------------------------------------------------------------------
// Stage 1: k_proj / q_proj GEMMs. 4 waves per block share one B panel (16 cols
// of one W1 half, all K=512), TDM-loaded into LDS once and reused 4x16 times.
// Steady state per WMMA: 2x global_load_b128 (A) + 32B LDS read (B).
// ---------------------------------------------------------------------------
__global__ __launch_bounds__(128) void proj_gemm_kernel(
    const _Float16* __restrict__ Ak, const _Float16* __restrict__ Aq,
    const _Float16* __restrict__ Whp, float* __restrict__ kp, float* __restrict__ qp)
{
  __shared__ _Float16 sB[16 * 32 * 16];            // 16 KB panel: [kk32][lane][16]

  const int bid  = blockIdx.x;
  const int wid  = threadIdx.x >> 5;
  const int lane = threadIdx.x & 31;
  const int l  = lane & 15;
  const int hi = lane >> 4;

  int wh, mg, t;
  const _Float16* A; float* dst;
  if (bid < 1024) { wh = 0; mg = bid >> 5;          t = bid & 31;          A = Ak; dst = kp; }
  else            { wh = 1; mg = (bid - 1024) >> 5; t = (bid - 1024) & 31; A = Aq; dst = qp; }

  const _Float16* panel = Whp + (size_t)(wh * 32 + t) * 8192;
#if HAVE_TDM
  if (wid == 0) {
    tdm_load_1d(panel, sB, 16384 >> 2);
    __builtin_amdgcn_s_wait_tensorcnt(0);
  }
#else
  coop_copy16(panel, sB, 16384);
#endif
  __syncthreads();

  const int tm = mg * 64 + wid * 16;
  const _Float16* arow = A + (size_t)(tm + l) * NHID;
  const int aofs = hi << 4;

  v8f c = {};
#pragma unroll
  for (int kk32 = 0; kk32 < 16; ++kk32) {
    const int kk = kk32 << 5;
    v8h a0 = *(const v8h*)(arow + kk + aofs);
    v8h a1 = *(const v8h*)(arow + kk + 8 + aofs);
    v16h a;
#pragma unroll
    for (int j = 0; j < 8; ++j) { a[j] = a0[j]; a[8 + j] = a1[j]; }
    v16h bv = *(const v16h*)(sB + ((kk32 << 5) + lane) * 16);
    c = __builtin_amdgcn_wmma_f32_16x16x32_f16(false, a, false, bv, (short)0, c, false, false);
  }

  float* drow = dst + (size_t)(tm + (hi << 3)) * NHID + t * 16 + l;
#pragma unroll
  for (int r = 0; r < 8; ++r) drow[(size_t)r * NHID] = c[r];
}

// ---------------------------------------------------------------------------
// Stage 2: fused score + mask + softmax per (q,b). VALU/tanh bound by design:
// score[k] = W2 . tanh(kp[k,b,:] + qp[q,b,:] + b1) + b2, softmax over k.
// Writes att_prob to d_out ([k,q,b]) and f16 [b,q,k] copy for stage 3 A-operand.
// ---------------------------------------------------------------------------
__global__ __launch_bounds__(256) void score_softmax_kernel(
    const float* __restrict__ kp, const float* __restrict__ qp,
    const float* __restrict__ b1, const float* __restrict__ W2,
    const float* __restrict__ b2, const unsigned char* __restrict__ mask,
    float* __restrict__ prob_out, _Float16* __restrict__ probTh)
{
  const int q = blockIdx.x >> 3;
  const int b = blockIdx.x & 7;

  __shared__ float s_q[NHID];
  __shared__ float s_w2[NHID];
  __shared__ float s_sc[TK];
  __shared__ float s_red[8];

  const int tid  = threadIdx.x;
  const int wv   = tid >> 5;
  const int lane = tid & 31;

  const float* qrow = qp + (size_t)(q * BB + b) * NHID;
  for (int h = tid; h < NHID; h += 256) { s_q[h] = qrow[h] + b1[h]; s_w2[h] = W2[h]; }
  __syncthreads();

  const float bias2 = b2[0];
  for (int k = wv; k < TK; k += 8) {
    const float* krow = kp + (size_t)(k * BB + b) * NHID;
    if (k + 8 < TK) __builtin_prefetch(kp + (size_t)((k + 8) * BB + b) * NHID, 0, 3);
    float p = 0.f;
    for (int h = lane; h < NHID; h += 32)
      p += tanhf(krow[h] + s_q[h]) * s_w2[h];
#pragma unroll
    for (int off = 16; off; off >>= 1) p += __shfl_xor(p, off, 32);
    if (lane == 0) s_sc[k] = mask[k * BB + b] ? -INFINITY : (p + bias2);
  }
  __syncthreads();

  float v = s_sc[tid];
  float m = v;
#pragma unroll
  for (int off = 16; off; off >>= 1) m = fmaxf(m, __shfl_xor(m, off, 32));
  if (lane == 0) s_red[wv] = m;
  __syncthreads();
  m = s_red[0];
#pragma unroll
  for (int i = 1; i < 8; ++i) m = fmaxf(m, s_red[i]);

  float e = expf(v - m);
  float s = e;
#pragma unroll
  for (int off = 16; off; off >>= 1) s += __shfl_xor(s, off, 32);
  __syncthreads();
  if (lane == 0) s_red[wv] = s;
  __syncthreads();
  s = 0.f;
#pragma unroll
  for (int i = 0; i < 8; ++i) s += s_red[i];

  const float pr = e / s;
  prob_out[((size_t)tid * TQ + q) * BB + b] = pr;                 // [k,q,b]
  probTh[((size_t)b * TQ + q) * TK + tid] = (_Float16)pr;         // [b,q,k] f16
}

// ---------------------------------------------------------------------------
// Stage 3: ant_vec = prob^T @ c_key per batch. Same TDM panel scheme, K=256.
// ---------------------------------------------------------------------------
__global__ __launch_bounds__(128) void av_gemm_kernel(
    const _Float16* __restrict__ probTh, const _Float16* __restrict__ ckp,
    float* __restrict__ ant)
{
  __shared__ _Float16 sB[8 * 32 * 16];             // 8 KB panel: [kk32][lane][16]

  const int bid  = blockIdx.x;
  const int wid  = threadIdx.x >> 5;
  const int lane = threadIdx.x & 31;
  const int l  = lane & 15;
  const int hi = lane >> 4;

  const int b  = bid >> 6;          // 64 blocks per batch
  const int r  = bid & 63;
  const int t  = r >> 1;            // h tile (32)
  const int mg = r & 1;             // q tile group (2 x 64 rows)

  const _Float16* panel = ckp + (size_t)(b * 32 + t) * 4096;
#if HAVE_TDM
  if (wid == 0) {
    tdm_load_1d(panel, sB, 8192 >> 2);
    __builtin_amdgcn_s_wait_tensorcnt(0);
  }
#else
  coop_copy16(panel, sB, 8192);
#endif
  __syncthreads();

  const int tm = mg * 64 + wid * 16;
  const _Float16* arow = probTh + ((size_t)b * TQ + tm + l) * TK;
  const int aofs = hi << 4;

  v8f c = {};
#pragma unroll
  for (int kk32 = 0; kk32 < 8; ++kk32) {
    const int kk = kk32 << 5;
    v8h a0 = *(const v8h*)(arow + kk + aofs);
    v8h a1 = *(const v8h*)(arow + kk + 8 + aofs);
    v16h a;
#pragma unroll
    for (int j = 0; j < 8; ++j) { a[j] = a0[j]; a[8 + j] = a1[j]; }
    v16h bv = *(const v16h*)(sB + ((kk32 << 5) + lane) * 16);
    c = __builtin_amdgcn_wmma_f32_16x16x32_f16(false, a, false, bv, (short)0, c, false, false);
  }

#pragma unroll
  for (int rr = 0; rr < 8; ++rr)
    ant[((size_t)(tm + (hi << 3) + rr) * BB + b) * NHID + t * 16 + l] = c[rr];
}

// ---------------------------------------------------------------------------
extern "C" void kernel_launch(void* const* d_in, const int* in_sizes, int n_in,
                              void* d_out, int out_size, void* d_ws, size_t ws_size,
                              hipStream_t stream) {
  (void)in_sizes; (void)n_in; (void)out_size; (void)ws_size;
  const float*         query = (const float*)d_in[0];        // [128,8,512]
  const float*         c_key = (const float*)d_in[1];        // [256,8,512]
  const unsigned char* mask  = (const unsigned char*)d_in[2];// [256,8] bool
  const float*         W1    = (const float*)d_in[3];        // [1024,512]
  const float*         b1    = (const float*)d_in[4];        // [512]
  const float*         W2    = (const float*)d_in[5];        // [512,1]
  const float*         b2    = (const float*)d_in[6];        // [1]

  float* out  = (float*)d_out;
  float* ant  = out;                                // [128,8,512] = 524288 f32
  float* prob = out + (size_t)TQ * BB * NHID;       // [256,128,8] = 262144 f32

  char* ws = (char*)d_ws;
  float*    kp    = (float*)(ws);                          // 4.0 MB  [2048,512] f32
  float*    qp    = (float*)(ws + (4u << 20));             // 2.0 MB  [1024,512] f32
  _Float16* ckh   = (_Float16*)(ws + (6u << 20));          // 2.0 MB  c_key f16 rows
  _Float16* qh    = (_Float16*)(ws + (8u << 20));          // 1.0 MB  query f16 rows
  _Float16* whp   = (_Float16*)(ws + (9u << 20));          // 1.0 MB  W1 B-image
  _Float16* ckp   = (_Float16*)(ws + (10u << 20));         // 2.0 MB  c_key B-image
  _Float16* pth   = (_Float16*)(ws + (12u << 20));         // 0.5 MB  probT f16

  cvt_f16_kernel <<<(2048 * NHID) / 256, 256, 0, stream>>>(c_key, ckh, 2048 * NHID);
  cvt_f16_kernel <<<(1024 * NHID) / 256, 256, 0, stream>>>(query, qh, 1024 * NHID);
  pack_w_kernel  <<<524288 / 256, 256, 0, stream>>>(W1, whp);
  pack_ck_kernel <<<1048576 / 256, 256, 0, stream>>>(c_key, ckp);

  proj_gemm_kernel     <<<1536, 128, 0, stream>>>(ckh, qh, whp, kp, qp);
  score_softmax_kernel <<<TQ * BB, 256, 0, stream>>>(kp, qp, b1, W2, b2, mask, prob, pth);
  av_gemm_kernel       <<<512, 128, 0, stream>>>(pth, ckp, ant);
}
